// MultiHeadScaledDotProductAttention_50603304681542
// MI455X (gfx1250) — matrix operand order repaired and checked by reading, not compiled, not run on hardware
//
#include <hip/hip_runtime.h>
#include <stdint.h>
#include <math.h>

typedef __bf16 bf16;
typedef __attribute__((ext_vector_type(16))) __bf16 v16bf;
typedef __attribute__((ext_vector_type(8)))  __bf16 v8bf;
typedef __attribute__((ext_vector_type(8)))  float  v8f;
typedef __attribute__((ext_vector_type(4)))  float  v4f;

static constexpr int Bn = 4;
static constexpr int Nn = 2048;
static constexpr int Dn = 128;
static constexpr int Hn = 10;
static constexpr int HD = Hn * Dn; // 1280

union AFrag { v16bf v; v8bf h2[2]; bf16 e[16]; };

__device__ __forceinline__ v8f wmma_bf16(v16bf a, v16bf b, v8f c) {
    return __builtin_amdgcn_wmma_f32_16x16x32_bf16(false, a, false, b, (short)0, c, false, false);
}

__device__ __forceinline__ float half_max(float v) {
    #pragma unroll
    for (int off = 1; off < 16; off <<= 1) v = fmaxf(v, __shfl_xor(v, off, 32));
    return v;
}
__device__ __forceinline__ float half_sum(float v) {
    #pragma unroll
    for (int off = 1; off < 16; off <<= 1) v += __shfl_xor(v, off, 32);
    return v;
}

// ---- CDNA5 async global->LDS path (guarded; fallback = vectorized manual copy)
#if defined(__AMDGCN__) && __has_builtin(__builtin_amdgcn_global_load_async_to_lds_b128)
#define HAS_ASYNC_LDS 1
#endif

#ifdef HAS_ASYNC_LDS
typedef int v4i __attribute__((vector_size(16)));            // matches builtin pointee type
typedef __attribute__((address_space(1))) v4i gv4i;          // global (device) int4
typedef __attribute__((address_space(3))) v4i lv4i;          // LDS int4
__device__ __forceinline__ gv4i* as_global16(const void* p) {
    return (gv4i*)(uintptr_t)p;
}
__device__ __forceinline__ lv4i* as_lds16(void* p) {
    return (lv4i*)(uint32_t)(uintptr_t)p;   // generic LDS addr: low 32 bits = LDS offset
}
#endif

__device__ __forceinline__ void wait_async0() {
#if defined(__AMDGCN__)
#if __has_builtin(__builtin_amdgcn_s_wait_asynccnt)
    __builtin_amdgcn_s_wait_asynccnt(0);
#else
    asm volatile("s_wait_asynccnt 0" ::: "memory");
#endif
#endif
}

// ---------------- Kernel 0: one-shot f32 -> bf16 conversion ----------------
__global__ __launch_bounds__(256)
void cvt_kernel(const float* __restrict__ in, bf16* __restrict__ out, int n4) {
    const int i = blockIdx.x * 256 + threadIdx.x;  // index over float4 groups
    if (i < n4) {
        const v4f v = *(const v4f*)(in + (size_t)i * 4);
        bf16* o = out + (size_t)i * 4;
        o[0] = (bf16)v.x; o[1] = (bf16)v.y; o[2] = (bf16)v.z; o[3] = (bf16)v.w;
    }
}

// ---------------- Kernel 1: QKV projections (bf16 in/out, f32 accum) --------
// grid: (B*N/16, H, 3), block: 128 (4 waves, each owns 2 e-tiles of 16)
__global__ __launch_bounds__(128)
void qkv_proj_kernel(const bf16* __restrict__ xbf,
                     const bf16* __restrict__ Wqb,
                     const bf16* __restrict__ Wkb,
                     const bf16* __restrict__ Wvb,
                     bf16* __restrict__ Qo, bf16* __restrict__ Ko, bf16* __restrict__ Vo)
{
    const int lane  = threadIdx.x & 31;
    const int wave  = threadIdx.x >> 5;
    const int lhalf = lane & 15;
    const int hi    = lane >> 4;
    const int row0  = blockIdx.x * 16;           // flat row in [0, B*N)
    const int h     = blockIdx.y;
    const int which = blockIdx.z;

    const bf16* W = (which == 0) ? Wqb : (which == 1) ? Wkb : Wvb;
    bf16* Out     = (which == 0) ? Qo  : (which == 1) ? Ko  : Vo;
    W += (size_t)h * Dn * Dn;

    const int b  = row0 / Nn;
    const int n0 = row0 - b * Nn;

    // A fragments for all 4 k-chunks (shared by both e-tiles of this wave)
    AFrag A[4];
    {
        const bf16* xr = xbf + (size_t)(row0 + lhalf) * Dn + 16 * hi;
        #pragma unroll
        for (int c = 0; c < 4; ++c) {
            A[c].v = *(const v16bf*)(xr + 32 * c);
        }
    }

    #pragma unroll
    for (int tt = 0; tt < 2; ++tt) {
        const int et = wave * 2 + tt;            // e-tile 0..7
        const bf16* wr = W + (size_t)(et * 16 + lhalf) * Dn + 16 * hi;
        v8f acc = {};
        #pragma unroll
        for (int c = 0; c < 4; ++c) {
            const v16bf Bv = *(const v16bf*)(wr + 32 * c);   // 32B aligned
            acc = wmma_bf16(A[c].v, Bv, acc);
        }
        // store C: lane<16 VGPR j -> (M=j, N=lhalf); lane>=16 -> (M=j+8, N=lhalf)
        #pragma unroll
        for (int j = 0; j < 8; ++j) {
            const int n = n0 + j + 8 * hi;
            Out[((size_t)(b * Hn + h) * Nn + n) * Dn + et * 16 + lhalf] = (bf16)acc[j];
        }
    }
}

// ---------------- Kernel 2: flash attention ----------------
// grid: (N/64, B*H), block: 128 (4 waves, each owns a 16-row query tile)
__global__ __launch_bounds__(128)
void attn_kernel(const bf16* __restrict__ Q, const bf16* __restrict__ K,
                 const bf16* __restrict__ V, bf16* __restrict__ O /* [B,N,H*D] */)
{
    __shared__ alignas(32) bf16 s_k [32 * 144];   // K tile row-major, padded (288B rows)
    __shared__ alignas(32) bf16 s_vt[128 * 48];   // V tile transposed [e][row], padded (96B rows)
    __shared__ alignas(32) bf16 s_p [4 * 16 * 48];// per-wave P staging

    const int tid   = threadIdx.x;
    const int lane  = tid & 31;
    const int wave  = tid >> 5;
    const int lhalf = lane & 15;
    const int hi    = lane >> 4;
    const int bh    = blockIdx.y;
    const int b     = bh / Hn, h = bh % Hn;
    const int qbase = blockIdx.x * 64 + wave * 16;

    const bf16* Qb = Q + (size_t)bh * Nn * Dn;
    const bf16* Kb = K + (size_t)bh * Nn * Dn;
    const bf16* Vb = V + (size_t)bh * Nn * Dn;

    // Q A-fragments (16 rows x 128 K), 4 chunks of K=32
    AFrag qa[4];
    {
        const bf16* qr = Qb + (size_t)(qbase + lhalf) * Dn + 16 * hi;
        #pragma unroll
        for (int c = 0; c < 4; ++c) {
            qa[c].v = *(const v16bf*)(qr + 32 * c);
        }
    }

    v8f acc[8];
    #pragma unroll
    for (int t = 0; t < 8; ++t) acc[t] = (v8f){};
    float m[8], l[8];
    #pragma unroll
    for (int j = 0; j < 8; ++j) { m[j] = -INFINITY; l[j] = 0.0f; }

    const float rscale = 1.0f / (sqrtf((float)Dn) + 1e-8f);
    bf16* s_pw = s_p + wave * 16 * 48;

    for (int kt = 0; kt < Nn / 32; ++kt) {
        const int kb = kt * 32;
        __syncthreads();

        // ---- stage K tile: 32x128 bf16, row stride 144, via async b128 copies
#ifdef HAS_ASYNC_LDS
        #pragma unroll
        for (int i = 0; i < 4; ++i) {
            const int idx = i * 128 + tid;       // 512 x 16B transfers
            const int r   = idx >> 4;
            const int seg = idx & 15;
            __builtin_amdgcn_global_load_async_to_lds_b128(
                as_global16(Kb + (size_t)(kb + r) * Dn + seg * 8),
                as_lds16(s_k + r * 144 + seg * 8), 0, 0);
        }
#else
        #pragma unroll
        for (int i = 0; i < 4; ++i) {
            const int idx = i * 128 + tid;
            const int r   = idx >> 4;
            const int seg = idx & 15;
            *(v8bf*)(s_k + r * 144 + seg * 8) =
                *(const v8bf*)(Kb + (size_t)(kb + r) * Dn + seg * 8);
        }
#endif
        // ---- stage V tile transposed: s_vt[e*48 + r] = V[r][e]
        #pragma unroll
        for (int i = 0; i < 2; ++i) {
            const int pidx = i * 128 + tid;      // 256 row-pair segments
            const int rp   = pidx >> 4;          // row pair 0..15
            const int seg  = pidx & 15;
            const v8bf va = *(const v8bf*)(Vb + (size_t)(kb + 2 * rp)     * Dn + seg * 8);
            const v8bf vb = *(const v8bf*)(Vb + (size_t)(kb + 2 * rp + 1) * Dn + seg * 8);
            #pragma unroll
            for (int k = 0; k < 8; ++k) {
                const int ei = (k + seg) & 7;    // rotate phase: avoid LDS bank conflicts
                union { bf16 p[2]; uint32_t u; } pk;
                pk.p[0] = va[ei]; pk.p[1] = vb[ei];
                *(uint32_t*)(s_vt + (seg * 8 + ei) * 48 + 2 * rp) = pk.u;
            }
        }
#ifdef HAS_ASYNC_LDS
        wait_async0();
#endif
        __syncthreads();

        // prefetch next K/V tiles into cache while we compute on this one
        if (kt + 1 < Nn / 32) {
            const char* nk = (const char*)(Kb + (size_t)(kb + 32) * Dn);
            const char* nv = (const char*)(Vb + (size_t)(kb + 32) * Dn);
            __builtin_prefetch(nk + tid * 64, 0, 0);
            __builtin_prefetch(nv + tid * 64, 0, 0);
        }

        // ---- S = Q * K^T for 32 key columns -> two 16x16 tiles
        v8f s0 = {}, s1 = {};
        #pragma unroll
        for (int c = 0; c < 4; ++c) {
            const v16bf b0 = *(const v16bf*)(s_k + (lhalf) * 144      + 32 * c + 16 * hi);
            const v16bf b1 = *(const v16bf*)(s_k + (16 + lhalf) * 144 + 32 * c + 16 * hi);
            s0 = wmma_bf16(qa[c].v, b0, s0);
            s1 = wmma_bf16(qa[c].v, b1, s1);
        }

        // ---- online softmax (row r = j + 8*hi lives in one 16-lane half)
        #pragma unroll
        for (int j = 0; j < 8; ++j) {
            const float a0 = s0[j] * rscale;
            const float a1 = s1[j] * rscale;
            float mx = half_max(fmaxf(a0, a1));
            const float mn    = fmaxf(m[j], mx);
            const float alpha = __expf(m[j] - mn);
            const float p0 = __expf(a0 - mn);
            const float p1 = __expf(a1 - mn);
            l[j] = l[j] * alpha + half_sum(p0 + p1);
            m[j] = mn;
            #pragma unroll
            for (int t = 0; t < 8; ++t) acc[t][j] *= alpha;
            const int prow = j + 8 * hi;
            s_pw[prow * 48 + lhalf]      = (bf16)p0;
            s_pw[prow * 48 + 16 + lhalf] = (bf16)p1;
        }

        // ---- reshape P (C layout) -> A fragment via per-wave LDS staging
        AFrag pa;
        {
            const bf16* pr = s_pw + lhalf * 48 + 16 * hi;
            pa.v = *(const v16bf*)(pr);
        }

        // ---- O += P * V  (8 column tiles of 16)
        #pragma unroll
        for (int t = 0; t < 8; ++t) {
            const v16bf bv = *(const v16bf*)(s_vt + (t * 16 + lhalf) * 48 + 16 * hi);
            acc[t] = wmma_bf16(pa.v, bv, acc[t]);
        }
    }

    // epilogue: normalize, store concat-head bf16 [B, N, H*D]
    #pragma unroll
    for (int j = 0; j < 8; ++j) {
        const float inv = 1.0f / l[j];
        const int row = qbase + j + 8 * hi;
        bf16* op = O + ((size_t)b * Nn + row) * HD + h * Dn;
        #pragma unroll
        for (int t = 0; t < 8; ++t) op[t * 16 + lhalf] = (bf16)(acc[t][j] * inv);
    }
}

// ---------------- Kernel 3: output projection ----------------
// grid: (B*N/16), block: 128 (4 waves x 2 e-tiles)
__global__ __launch_bounds__(128)
void out_proj_kernel(const bf16* __restrict__ Oc /* [B*N, H*D] */,
                     const bf16* __restrict__ Wob /* [D, H*D] bf16 */,
                     const float* __restrict__ bo,
                     float* __restrict__ out /* [B*N, D] */)
{
    const int lane  = threadIdx.x & 31;
    const int wave  = threadIdx.x >> 5;
    const int lhalf = lane & 15;
    const int hi    = lane >> 4;
    const int row0  = blockIdx.x * 16;
    const int et0   = wave * 2;

    v8f acc0 = {}, acc1 = {};
    const bf16* ar0 = Oc  + (size_t)(row0 + lhalf) * HD + 16 * hi;
    const bf16* w0  = Wob + (size_t)(et0 * 16 + lhalf) * HD + 16 * hi;
    const bf16* w1  = w0  + (size_t)16 * HD;
    #pragma unroll 4
    for (int c = 0; c < HD / 32; ++c) {    // 40 chunks
        AFrag a;
        a.v = *(const v16bf*)(ar0 + 32 * c);
        const v16bf b0 = *(const v16bf*)(w0 + 32 * c);
        const v16bf b1 = *(const v16bf*)(w1 + 32 * c);
        acc0 = wmma_bf16(a.v, b0, acc0);
        acc1 = wmma_bf16(a.v, b1, acc1);
    }
    const float bb0 = bo[et0 * 16 + lhalf];
    const float bb1 = bo[(et0 + 1) * 16 + lhalf];
    #pragma unroll
    for (int j = 0; j < 8; ++j) {
        const size_t r = (size_t)(row0 + j + 8 * hi) * Dn;
        out[r + et0 * 16 + lhalf]       = acc0[j] + bb0;
        out[r + (et0 + 1) * 16 + lhalf] = acc1[j] + bb1;
    }
}

extern "C" void kernel_launch(void* const* d_in, const int* in_sizes, int n_in,
                              void* d_out, int out_size, void* d_ws, size_t ws_size,
                              hipStream_t stream) {
    (void)in_sizes; (void)n_in; (void)out_size; (void)ws_size;
    const float* x  = (const float*)d_in[0];
    const float* Wq = (const float*)d_in[1];
    const float* Wk = (const float*)d_in[2];
    const float* Wv = (const float*)d_in[3];
    const float* Wo = (const float*)d_in[4];
    const float* bo = (const float*)d_in[5];

    const size_t nX   = (size_t)Bn * Nn * Dn;        // 1,048,576
    const size_t nW   = (size_t)Hn * Dn * Dn;        //   163,840
    const size_t nWo  = (size_t)Dn * HD;             //   163,840
    const size_t per  = (size_t)Bn * Hn * Nn * Dn;   // 10,485,760

    bf16* xbf = (bf16*)d_ws;
    bf16* wqb = xbf + nX;
    bf16* wkb = wqb + nW;
    bf16* wvb = wkb + nW;
    bf16* wob = wvb + nW;
    bf16* qbf = wob + nWo;
    bf16* kbf = qbf + per;
    bf16* vbf = kbf + per;
    bf16* obf = vbf + per;                            // [B*N, H*D] concat

    auto cvt = [&](const float* src, bf16* dst, size_t n) {
        const int n4 = (int)(n / 4);
        cvt_kernel<<<dim3((n4 + 255) / 256), 256, 0, stream>>>(src, dst, n4);
    };
    cvt(x,  xbf, nX);
    cvt(Wq, wqb, nW);
    cvt(Wk, wkb, nW);
    cvt(Wv, wvb, nW);
    cvt(Wo, wob, nWo);

    qkv_proj_kernel<<<dim3(Bn * Nn / 16, Hn, 3), 128, 0, stream>>>(
        xbf, wqb, wkb, wvb, qbf, kbf, vbf);
    attn_kernel<<<dim3(Nn / 64, Bn * Hn), 128, 0, stream>>>(qbf, kbf, vbf, obf);
    out_proj_kernel<<<dim3(Bn * Nn / 16), 128, 0, stream>>>(obf, wob, bo, (float*)d_out);
}
